// Tacotron2Model_18442589570238
// MI455X (gfx1250) — compile-verified, matrix operand order checked by reading, not run on hardware
//
#include <hip/hip_runtime.h>

// ---------------------------------------------------------------------------
// Types for WMMA
// ---------------------------------------------------------------------------
typedef _Float16 f16;
typedef __attribute__((ext_vector_type(16))) _Float16 v16h;
typedef __attribute__((ext_vector_type(8)))  _Float16 v8h;
typedef __attribute__((ext_vector_type(8)))  float    v8f;

#define B_   8
#define S_   128
#define H_   512
#define MEL_ 80

__device__ __forceinline__ v8f wmma32(v16h a, v16h b, v8f c) {
  // D = A(16x32 f16) * B(32x16 f16) + C(16x16 f32)
  return __builtin_amdgcn_wmma_f32_16x16x32_f16(
      /*neg_a=*/false, a, /*neg_b=*/false, b,
      /*c_mod=*/(short)0, c, /*reuse_a=*/false, /*reuse_b=*/false);
}

// Load one 16x32 f16 operand tile.
// A-matrix striping (ISA 7.12.2, 16-bit 16x32): lane<16 holds row M=lane,
// K={k0..k0+7} and {k0+16..k0+23}; lane>=16 holds row M=lane-16,
// K={k0+8..k0+15} and {k0+24..k0+31}.  B (as W^T, W row-major [N][K]) uses the
// mirrored layout with N across lanes, so the same loader serves both with
// base = &M[row0][k0] (or &W[n0][k0]) and ld = row stride in elements.
__device__ __forceinline__ v16h ld_tile(const f16* __restrict__ base, int ld) {
  const int lane = threadIdx.x & 31;
  const f16* p = base + (lane & 15) * ld + ((lane >> 4) << 3);
  v8h lo = *reinterpret_cast<const v8h*>(p);       // K j..j+7
  v8h hi = *reinterpret_cast<const v8h*>(p + 16);  // K j+16..j+23
  v16h r;
#pragma unroll
  for (int i = 0; i < 8; ++i) { r[i] = lo[i]; r[i + 8] = hi[i]; }
  return r;
}

__device__ __forceinline__ float sigm(float x) { return 1.f / (1.f + __expf(-x)); }

// ---------------------------------------------------------------------------
// Elementwise helpers
// ---------------------------------------------------------------------------
__global__ void cvt16_k(const float* __restrict__ src, int srcLd, int colOff,
                        f16* __restrict__ dst, int cols, int dstLd, int total) {
  int idx = blockIdx.x * blockDim.x + threadIdx.x;
  if (idx >= total) return;
  int r = idx / dstLd, cp = idx - r * dstLd;
  float v = (cp < cols) ? src[(long)r * srcLd + colOff + cp] : 0.f;
  dst[idx] = (f16)v;
}

__global__ void embed_k(const int* __restrict__ text, const float* __restrict__ emb,
                        f16* __restrict__ x16) {
  int idx = blockIdx.x * blockDim.x + threadIdx.x;   // over B*S*H
  if (idx >= B_ * S_ * H_) return;
  int row = idx >> 9, k = idx & (H_ - 1);
  x16[idx] = (f16)emb[text[row] * H_ + k];
}

// ---------------------------------------------------------------------------
// Parallel GEMM:  C[M][N] (f32) = A16[M][K] @ W16[N][K]^T + b1[n] + b2[n]
// One wave per 16x16 tile; block = 256 threads = 8 tiles.
// ---------------------------------------------------------------------------
__global__ __launch_bounds__(256) void gemm_bias_k(
    const f16* __restrict__ A, const f16* __restrict__ W,
    const float* __restrict__ b1, const float* __restrict__ b2,
    float* __restrict__ C, int M, int N, int K) {
  const int wave = threadIdx.x >> 5;
  const int lane = threadIdx.x & 31;
  const int tile = blockIdx.x * (blockDim.x >> 5) + wave;
  const int ntiles = N >> 4;
  if (tile >= (M >> 4) * ntiles) return;           // wave-uniform
  const int mt = tile / ntiles, nt = tile - mt * ntiles;
  v8f acc = {};
  const f16* Ab = A + (long)(mt << 4) * K;
  const f16* Wb = W + (long)(nt << 4) * K;
  for (int k = 0; k < K; k += 32) {
    v16h a = ld_tile(Ab + k, K);
    v16h b = ld_tile(Wb + k, K);
    acc = wmma32(a, b, acc);
  }
  const int col = (nt << 4) + (lane & 15);
  float bias = (b1 ? b1[col] : 0.f) + (b2 ? b2[col] : 0.f);
  const int rbase = (mt << 4) + ((lane >> 4) << 3);
#pragma unroll
  for (int r = 0; r < 8; ++r)
    C[(long)(rbase + r) * N + col] = acc[r] + bias;
}

// ---------------------------------------------------------------------------
// LSTM sequential scan. grid.x = 2 (dir 0 = forward, dir 1 = backward).
// Z* = precomputed X@Wih^T + bih + bhh, rows b*S+s, width 4H.
// Wave w owns hidden cols [16w,16w+16) and its 4 gate tiles; cell state c
// stays in accumulator-layout VGPRs for all 128 steps.
// ---------------------------------------------------------------------------
__global__ __launch_bounds__(1024) void lstm_scan_k(
    const float* __restrict__ Zf, const float* __restrict__ Zb,
    const f16* __restrict__ Wfh, const f16* __restrict__ Wbh,
    f16* __restrict__ out16, int outLd, float* __restrict__ outF32) {
  __shared__ f16 hsh[16][H_];
  const int dir  = blockIdx.x;
  const float* Z = dir ? Zb : Zf;
  const f16*   W = dir ? Wbh : Wfh;
  const int colOff = dir ? H_ : 0;
  const int wave = threadIdx.x >> 5;
  const int lane = threadIdx.x & 31;
  const int nb   = wave << 4;
  const int j    = nb + (lane & 15);

  for (int i = threadIdx.x; i < 16 * H_; i += 1024) ((f16*)hsh)[i] = (f16)0.f;
  __syncthreads();

  float c[8];
#pragma unroll
  for (int r = 0; r < 8; ++r) c[r] = 0.f;

  for (int s = 0; s < S_; ++s) {
    const int sIdx = dir ? (S_ - 1 - s) : s;
    v8f gi = {}, gf = {}, gg = {}, go = {};
    const f16* Wi = W + (long)(0 * H_ + nb) * H_;
    const f16* Wf2 = W + (long)(1 * H_ + nb) * H_;
    const f16* Wg = W + (long)(2 * H_ + nb) * H_;
    const f16* Wo = W + (long)(3 * H_ + nb) * H_;
    for (int k = 0; k < H_; k += 32) {
      v16h a = ld_tile(&hsh[0][0] + k, H_);
      __builtin_prefetch(Wi + k + 128, 0, 0);
      gi = wmma32(a, ld_tile(Wi + k, H_), gi);
      gf = wmma32(a, ld_tile(Wf2 + k, H_), gf);
      gg = wmma32(a, ld_tile(Wg + k, H_), gg);
      go = wmma32(a, ld_tile(Wo + k, H_), go);
    }
    __syncthreads();                 // all waves done reading hsh
    if (lane < 16) {
#pragma unroll
      for (int r = 0; r < 8; ++r) {
        const long row = (long)r * S_ + sIdx;    // b*S+s
        const float xi = gi[r] + Z[row * 2048 + 0 * H_ + j];
        const float xf = gf[r] + Z[row * 2048 + 1 * H_ + j];
        const float xg = gg[r] + Z[row * 2048 + 2 * H_ + j];
        const float xo = go[r] + Z[row * 2048 + 3 * H_ + j];
        const float c2 = sigm(xf) * c[r] + sigm(xi) * tanhf(xg);
        const float h2 = sigm(xo) * tanhf(c2);
        c[r] = c2;
        hsh[r][j] = (f16)h2;
        out16[row * outLd + colOff + j] = (f16)h2;
        if (outF32) outF32[row * outLd + colOff + j] = h2;
      }
    }
    __syncthreads();                 // hsh updated for next step
  }
}

// ---------------------------------------------------------------------------
// Attention decoder: one persistent workgroup, 1024 threads, T steps.
// ---------------------------------------------------------------------------
#define XINLD 1120                   // 1104 padded to multiple of 32

__global__ __launch_bounds__(1024) void decoder_k(
    const float* __restrict__ encPart,   // [B*S][H]
    const float* __restrict__ encF,      // [B*S][2H]
    const f16*   __restrict__ Wh16,      // [H][H]
    const float* __restrict__ attnV,     // [H]
    const f16*   __restrict__ dWih16,    // [4H][XINLD]
    const f16*   __restrict__ dWhh16,    // [4H][H]
    const float* __restrict__ dbih, const float* __restrict__ dbhh,
    const f16*   __restrict__ melW16,    // [MEL][H]
    const float* __restrict__ melB,
    float* __restrict__ out, int T) {
  extern __shared__ char smem[];
  f16   (*hsh)[H_]    = (f16(*)[H_])(smem);                    // 16384
  f16   (*xin)[XINLD] = (f16(*)[XINLD])(smem + 16384);         // 35840 -> 52224
  float (*hw)[H_]     = (float(*)[H_])(smem + 52224);          // 16384 -> 68608
  float (*score)[S_]  = (float(*)[S_])(smem + 68608);          // 4096  -> 72704
  float (*aw)[S_]     = (float(*)[S_])(smem + 72704);          // 4096  -> 76800
  float (*melp)[MEL_] = (float(*)[MEL_])(smem + 76800);        // 2560  -> 79360
  float *vsh          = (float*)(smem + 79360);                // 2048  -> 81408

  const int tid  = threadIdx.x;
  const int wave = tid >> 5, lane = tid & 31;
  const int nb   = wave << 4;
  const int j    = nb + (lane & 15);

  for (int i = tid; i < 16 * H_; i += 1024)    ((f16*)hsh)[i] = (f16)0.f;
  for (int i = tid; i < 16 * XINLD; i += 1024) ((f16*)xin)[i] = (f16)0.f;
  for (int i = tid; i < B_ * MEL_; i += 1024)  ((float*)melp)[i] = 0.f;
  for (int i = tid; i < H_; i += 1024)         vsh[i] = attnV[i];
  __syncthreads();

  float c[8];
#pragma unroll
  for (int r = 0; r < 8; ++r) c[r] = 0.f;

  for (int t = 0; t < T; ++t) {
    // ---- 1: hw = h @ W_h^T  (wave w -> col tile nb) -----------------------
    {
      v8f acc = {};
      const f16* Wb = Wh16 + (long)nb * H_;
      for (int k = 0; k < H_; k += 32) {
        v16h a = ld_tile(&hsh[0][0] + k, H_);
        acc = wmma32(a, ld_tile(Wb + k, H_), acc);
      }
      if (lane < 16) {
#pragma unroll
        for (int r = 0; r < 8; ++r) hw[r][j] = acc[r];
      }
    }
    __syncthreads();
    // ---- 2: score[b][s] = sum_k tanh(encPart + hw) * v --------------------
    {
      const int b = tid >> 7, s = tid & (S_ - 1);
      const float* ep = encPart + (long)(b * S_ + s) * H_;
      float sum = 0.f;
      for (int k = 0; k < H_; ++k) sum += tanhf(ep[k] + hw[b][k]) * vsh[k];
      score[b][s] = sum;
    }
    __syncthreads();
    // ---- 3: softmax over s ------------------------------------------------
    if (tid < B_) {
      float mx = -1e30f;
      for (int s = 0; s < S_; ++s) mx = fmaxf(mx, score[tid][s]);
      float sum = 0.f;
      for (int s = 0; s < S_; ++s) { float e = __expf(score[tid][s] - mx); aw[tid][s] = e; sum += e; }
      float inv = 1.f / sum;
      for (int s = 0; s < S_; ++s) aw[tid][s] *= inv;
    }
    __syncthreads();
    // ---- 4: ctx = a @ enc ; build xin = [mel_prev | ctx | pad] ------------
    for (int idx = tid; idx < B_ * 1024; idx += 1024) {
      const int b = idx >> 10, h = idx & 1023;
      const float* e = encF + (long)b * S_ * 1024 + h;
      float sum = 0.f;
      for (int s = 0; s < S_; ++s) sum += aw[b][s] * e[s * 1024];
      xin[b][MEL_ + h] = (f16)sum;
    }
    for (int idx = tid; idx < B_ * MEL_; idx += 1024)
      xin[idx / MEL_][idx % MEL_] = (f16)melp[idx / MEL_][idx % MEL_];
    __syncthreads();
    // ---- 5: decoder LSTM cell (4 gate tiles per wave) ---------------------
    {
      v8f gi = {}, gf = {}, gg = {}, go = {};
      const f16* Pi = dWih16 + (long)(0 * H_ + nb) * XINLD;
      const f16* Pf = dWih16 + (long)(1 * H_ + nb) * XINLD;
      const f16* Pg = dWih16 + (long)(2 * H_ + nb) * XINLD;
      const f16* Po = dWih16 + (long)(3 * H_ + nb) * XINLD;
      for (int k = 0; k < XINLD; k += 32) {
        v16h a = ld_tile(&xin[0][0] + k, XINLD);
        __builtin_prefetch(Pi + k + 128, 0, 0);
        gi = wmma32(a, ld_tile(Pi + k, XINLD), gi);
        gf = wmma32(a, ld_tile(Pf + k, XINLD), gf);
        gg = wmma32(a, ld_tile(Pg + k, XINLD), gg);
        go = wmma32(a, ld_tile(Po + k, XINLD), go);
      }
      const f16* Qi = dWhh16 + (long)(0 * H_ + nb) * H_;
      const f16* Qf = dWhh16 + (long)(1 * H_ + nb) * H_;
      const f16* Qg = dWhh16 + (long)(2 * H_ + nb) * H_;
      const f16* Qo = dWhh16 + (long)(3 * H_ + nb) * H_;
      for (int k = 0; k < H_; k += 32) {
        v16h a = ld_tile(&hsh[0][0] + k, H_);
        gi = wmma32(a, ld_tile(Qi + k, H_), gi);
        gf = wmma32(a, ld_tile(Qf + k, H_), gf);
        gg = wmma32(a, ld_tile(Qg + k, H_), gg);
        go = wmma32(a, ld_tile(Qo + k, H_), go);
      }
      __syncthreads();               // done reading hsh / xin
      if (lane < 16) {
#pragma unroll
        for (int r = 0; r < 8; ++r) {
          const float xi = gi[r] + dbih[0 * H_ + j] + dbhh[0 * H_ + j];
          const float xf = gf[r] + dbih[1 * H_ + j] + dbhh[1 * H_ + j];
          const float xg = gg[r] + dbih[2 * H_ + j] + dbhh[2 * H_ + j];
          const float xo = go[r] + dbih[3 * H_ + j] + dbhh[3 * H_ + j];
          const float c2 = sigm(xf) * c[r] + sigm(xi) * tanhf(xg);
          const float h2 = sigm(xo) * tanhf(c2);
          c[r] = c2;
          hsh[r][j] = (f16)h2;
        }
      }
    }
    __syncthreads();
    // ---- 6: mel = h @ mel_W^T + b  (waves 0..4, N=80) ---------------------
    if (wave < 5) {
      v8f m = {};
      const f16* Mb = melW16 + (long)nb * H_;
      for (int k = 0; k < H_; k += 32) {
        v16h a = ld_tile(&hsh[0][0] + k, H_);
        m = wmma32(a, ld_tile(Mb + k, H_), m);
      }
      if (lane < 16) {
        const int n = nb + lane;
        const float bb = melB[n];
#pragma unroll
        for (int r = 0; r < 8; ++r) {
          const float val = m[r] + bb;
          melp[r][n] = val;
          out[((long)r * MEL_ + n) * T + t] = val;
        }
      }
    }
    __syncthreads();
  }
}

// ---------------------------------------------------------------------------
// Host launcher
// ---------------------------------------------------------------------------
extern "C" void kernel_launch(void* const* d_in, const int* in_sizes, int n_in,
                              void* d_out, int out_size, void* d_ws, size_t ws_size,
                              hipStream_t stream) {
  (void)in_sizes; (void)n_in; (void)ws_size;
  const int*   text   = (const int*)d_in[0];
  const float* emb    = (const float*)d_in[2];
  const float* Wih00  = (const float*)d_in[3];
  const float* Whh00  = (const float*)d_in[4];
  const float* bih00  = (const float*)d_in[5];
  const float* bhh00  = (const float*)d_in[6];
  const float* Wih01  = (const float*)d_in[7];
  const float* Whh01  = (const float*)d_in[8];
  const float* bih01  = (const float*)d_in[9];
  const float* bhh01  = (const float*)d_in[10];
  const float* Wih10  = (const float*)d_in[11];
  const float* Whh10  = (const float*)d_in[12];
  const float* bih10  = (const float*)d_in[13];
  const float* bhh10  = (const float*)d_in[14];
  const float* Wih11  = (const float*)d_in[15];
  const float* Whh11  = (const float*)d_in[16];
  const float* bih11  = (const float*)d_in[17];
  const float* bhh11  = (const float*)d_in[18];
  const float* attnW  = (const float*)d_in[19];
  const float* attnB  = (const float*)d_in[20];
  const float* attnV  = (const float*)d_in[21];
  const float* decWih = (const float*)d_in[22];
  const float* decWhh = (const float*)d_in[23];
  const float* decBih = (const float*)d_in[24];
  const float* decBhh = (const float*)d_in[25];
  const float* melW   = (const float*)d_in[26];
  const float* melB   = (const float*)d_in[27];
  float* out = (float*)d_out;
  const int T = out_size / (B_ * MEL_);

  char* ws = (char*)d_ws;
  size_t off = 0;
  auto alloc = [&](size_t bytes) -> void* {
    void* p = ws + off;
    off = (off + bytes + 255) & ~(size_t)255;
    return p;
  };
  const int R = B_ * S_;                         // 1024 rows
  f16*   xemb16   = (f16*)alloc((size_t)R * H_ * 2);
  f16*   x1_16    = (f16*)alloc((size_t)R * 1024 * 2);
  f16*   enc16    = (f16*)alloc((size_t)R * 1024 * 2);
  float* encF     = (float*)alloc((size_t)R * 1024 * 4);
  float* encPart  = (float*)alloc((size_t)R * H_ * 4);
  float* Z0       = (float*)alloc((size_t)R * 2048 * 4);
  float* Z1       = (float*)alloc((size_t)R * 2048 * 4);
  f16* Wih00_16 = (f16*)alloc((size_t)2048 * 512 * 2);
  f16* Whh00_16 = (f16*)alloc((size_t)2048 * 512 * 2);
  f16* Wih01_16 = (f16*)alloc((size_t)2048 * 512 * 2);
  f16* Whh01_16 = (f16*)alloc((size_t)2048 * 512 * 2);
  f16* Wih10_16 = (f16*)alloc((size_t)2048 * 1024 * 2);
  f16* Whh10_16 = (f16*)alloc((size_t)2048 * 512 * 2);
  f16* Wih11_16 = (f16*)alloc((size_t)2048 * 1024 * 2);
  f16* Whh11_16 = (f16*)alloc((size_t)2048 * 512 * 2);
  f16* Wh16     = (f16*)alloc((size_t)512 * 512 * 2);
  f16* We16     = (f16*)alloc((size_t)512 * 1024 * 2);
  f16* dWih16   = (f16*)alloc((size_t)2048 * XINLD * 2);
  f16* dWhh16   = (f16*)alloc((size_t)2048 * 512 * 2);
  f16* melW16   = (f16*)alloc((size_t)MEL_ * 512 * 2);

  auto cvt = [&](const float* src, int srcLd, int colOff, f16* dst,
                 int rows, int cols, int dstLd) {
    int total = rows * dstLd;
    cvt16_k<<<(total + 255) / 256, 256, 0, stream>>>(src, srcLd, colOff, dst,
                                                     cols, dstLd, total);
  };
  cvt(Wih00, 512, 0, Wih00_16, 2048, 512, 512);
  cvt(Whh00, 512, 0, Whh00_16, 2048, 512, 512);
  cvt(Wih01, 512, 0, Wih01_16, 2048, 512, 512);
  cvt(Whh01, 512, 0, Whh01_16, 2048, 512, 512);
  cvt(Wih10, 1024, 0, Wih10_16, 2048, 1024, 1024);
  cvt(Whh10, 512, 0, Whh10_16, 2048, 512, 512);
  cvt(Wih11, 1024, 0, Wih11_16, 2048, 1024, 1024);
  cvt(Whh11, 512, 0, Whh11_16, 2048, 512, 512);
  cvt(attnW, 1536, 0,   Wh16, 512, 512, 512);
  cvt(attnW, 1536, 512, We16, 512, 1024, 1024);
  cvt(decWih, 1104, 0, dWih16, 2048, 1104, XINLD);
  cvt(decWhh, 512, 0, dWhh16, 2048, 512, 512);
  cvt(melW, 512, 0, melW16, MEL_, 512, 512);

  embed_k<<<(R * H_ + 255) / 256, 256, 0, stream>>>(text, emb, xemb16);

  auto gemm = [&](const f16* A, const f16* W, const float* b1, const float* b2,
                  float* C, int M, int N, int K) {
    int tiles = (M >> 4) * (N >> 4);
    gemm_bias_k<<<(tiles + 7) / 8, 256, 0, stream>>>(A, W, b1, b2, C, M, N, K);
  };

  // Layer 0: input projections then sequential scan (fwd+bwd concurrently)
  gemm(xemb16, Wih00_16, bih00, bhh00, Z0, R, 2048, 512);
  gemm(xemb16, Wih01_16, bih01, bhh01, Z1, R, 2048, 512);
  lstm_scan_k<<<2, 1024, 0, stream>>>(Z0, Z1, Whh00_16, Whh01_16,
                                      x1_16, 1024, nullptr);
  // Layer 1
  gemm(x1_16, Wih10_16, bih10, bhh10, Z0, R, 2048, 1024);
  gemm(x1_16, Wih11_16, bih11, bhh11, Z1, R, 2048, 1024);
  lstm_scan_k<<<2, 1024, 0, stream>>>(Z0, Z1, Whh10_16, Whh11_16,
                                      enc16, 1024, encF);
  // Attention precompute: enc_part = enc @ W_e^T + attn_b
  gemm(enc16, We16, attnB, nullptr, encPart, R, 512, 1024);

  // Decoder loop (persistent single workgroup)
  const size_t smemBytes = 81920;
  decoder_k<<<1, 1024, smemBytes, stream>>>(encPart, encF, Wh16, attnV,
                                            dWih16, dWhh16, decBih, decBhh,
                                            melW16, melB, out, T);
}
